// LlamaBlock_69922067579087
// MI455X (gfx1250) — matrix operand order repaired and checked by reading, not compiled, not run on hardware
//
#include <hip/hip_runtime.h>
#include <hip/hip_bf16.h>
#include <cstdint>
#include <cstddef>

#define DEV_INLINE __device__ __forceinline__

typedef __bf16 bf16_t;
typedef __attribute__((ext_vector_type(16))) __bf16 v16bf;
typedef __attribute__((ext_vector_type(8)))  float  v8f;
typedef __attribute__((ext_vector_type(4)))  unsigned int uint32x4;
typedef __attribute__((ext_vector_type(8)))  int  int32x8;
typedef __attribute__((ext_vector_type(4)))  int  int32x4;

constexpr int B_   = 2;
constexpr int T_   = 2048;
constexpr int C_   = 2048;
constexpr int H_   = 16;
constexpr int KV_  = 4;
constexpr int HD_  = 128;
constexpr int FFN_ = 5632;
constexpr int BT_  = B_ * T_;
constexpr int REP_ = H_ / KV_;

union Frag16 { v16bf v; uint4 u[2]; };

DEV_INLINE v8f wmma_bf16(v16bf a, v16bf b, v8f c) {
  // D = A(16x32) * B(32x16) + C, fp32 accumulate
  return __builtin_amdgcn_wmma_f32_16x16x32_bf16(false, a, false, b, (short)0, c,
                                                 false, false);
}

DEV_INLINE float half16_max(float v) {   // reduce across 16-lane half (C-layout row)
  v = fmaxf(v, __shfl_xor(v, 1, 32));
  v = fmaxf(v, __shfl_xor(v, 2, 32));
  v = fmaxf(v, __shfl_xor(v, 4, 32));
  v = fmaxf(v, __shfl_xor(v, 8, 32));
  return v;
}
DEV_INLINE float half16_sum(float v) {
  v += __shfl_xor(v, 1, 32);
  v += __shfl_xor(v, 2, 32);
  v += __shfl_xor(v, 4, 32);
  v += __shfl_xor(v, 8, 32);
  return v;
}

// ---------------------------------------------------------------------------
// Tensor Data Mover: 2D tile load Global -> LDS (D# per CDNA5 ISA)
//   tile: tile_k elements wide (data_size=2B), tile_rows tall
//   tensor: dim0 = Kdim elements, dim1 = Rows, row stride = Kdim elements
// ---------------------------------------------------------------------------
DEV_INLINE void tdm_load_2d(unsigned int lds_addr, const bf16_t* gptr,
                            int tile_k, int tile_rows, int Kdim, int Rows) {
  const unsigned long long ga = (unsigned long long)(size_t)gptr;
  uint32x4 g0;
  g0[0] = 1u;                                    // count=1, user mode, no gather
  g0[1] = lds_addr;                              // [63:32]  lds_addr (bytes)
  g0[2] = (unsigned int)ga;                      // [95:64]  global_addr lo
  g0[3] = (unsigned int)((ga >> 32) & 0x01FFFFFFu) | (2u << 30); // addr hi | type=2
  int32x8 g1;
  g1[0] = (int)(1u << 16);                       // workgroup_mask=0, data_size=1 (2B)
  g1[1] = (int)(((unsigned)Kdim & 0xFFFFu) << 16);          // tensor_dim0 [15:0]
  g1[2] = (int)(((unsigned)Kdim >> 16) |
                (((unsigned)Rows & 0xFFFFu) << 16));        // dim0 hi | dim1 lo
  g1[3] = (int)(((unsigned)Rows >> 16) |
                ((unsigned)tile_k << 16));                  // dim1 hi | tile_dim0
  g1[4] = (int)((unsigned)tile_rows);                       // tile_dim1 | tile_dim2=0
  g1[5] = (int)(unsigned)Kdim;                              // tensor_dim0_stride lo32
  g1[6] = 0;                                                // stride0 hi | stride1 lo
  g1[7] = 0;                                                // stride1 hi
  const int32x4 z4 = {0, 0, 0, 0};
#if __clang_major__ >= 23
  const int32x8 z8 = {0, 0, 0, 0, 0, 0, 0, 0};
  __builtin_amdgcn_tensor_load_to_lds(g0, g1, z4, z4, z8, 0);
#else
  __builtin_amdgcn_tensor_load_to_lds(g0, g1, z4, z4, 0);
#endif
}

// ---------------------------------------------------------------------------
// fp32 -> bf16 conversion
// ---------------------------------------------------------------------------
__global__ void cvt_f32_bf16(const float* __restrict__ in, bf16_t* __restrict__ out,
                             size_t n) {
  size_t i = (size_t)blockIdx.x * blockDim.x + threadIdx.x;
  if (i < n) out[i] = (bf16_t)in[i];
}

// ---------------------------------------------------------------------------
// RMSNorm: one block per row of length C_, bf16 output
// ---------------------------------------------------------------------------
__global__ __launch_bounds__(256) void rmsnorm_bf16(const float* __restrict__ x,
                                                    const float* __restrict__ g,
                                                    bf16_t* __restrict__ out) {
  const int row = blockIdx.x;
  const float* xr = x + (size_t)row * C_;
  float ss = 0.f;
  for (int i = threadIdx.x; i < C_; i += 256) { float v = xr[i]; ss += v * v; }
  for (int m = 16; m; m >>= 1) ss += __shfl_xor(ss, m, 32);
  __shared__ float red[8];
  if ((threadIdx.x & 31) == 0) red[threadIdx.x >> 5] = ss;
  __syncthreads();
  float tot = 0.f;
  for (int i = 0; i < 8; ++i) tot += red[i];
  const float rinv = rsqrtf(tot * (1.0f / C_) + 1e-5f);
  for (int i = threadIdx.x; i < C_; i += 256)
    out[(size_t)row * C_ + i] = (bf16_t)(xr[i] * rinv * g[i]);
}

// ---------------------------------------------------------------------------
// GEMM: Out[M,N] = A[M,K] (bf16) x W[N,K]^T (bf16).
// Block tile 128x64, 8 waves; wave tile 32x32 (4 WMMA accums).
// A/B tiles staged by the Tensor Data Mover, double-buffered (TENSORcnt).
// EPI 0: bf16 store | EPI 1: f32 store = acc + Res | EPI 2: f32 Out += acc
// ---------------------------------------------------------------------------
template <int EPI>
__global__ __launch_bounds__(256) void gemm_bf16_wmma(const bf16_t* __restrict__ A,
                                                      const bf16_t* __restrict__ W,
                                                      void* __restrict__ Out,
                                                      const float* __restrict__ Res,
                                                      int M, int N, int K) {
  __shared__ __align__(16) bf16_t lA[2][128 * 32];
  __shared__ __align__(16) bf16_t lB[2][64 * 32];

  const int tid  = threadIdx.x;
  const int lane = tid & 31;
  const int wave = tid >> 5;       // 0..7
  const int wm   = wave & 3;       // M sub-tile (32 rows each)
  const int wn   = wave >> 2;      // N sub-tile (32 cols each)
  const int bm   = blockIdx.y * 128;
  const int bn   = blockIdx.x * 64;

  const int am   = lane & 15;      // A fragment row within 16
  const int kh   = lane >> 4;      // A fragment K-half selector
  const int cn   = lane & 15;      // B/C fragment column
  const int half = lane >> 4;

  const unsigned int ldsA[2] = {(unsigned int)(size_t)&lA[0][0],
                                (unsigned int)(size_t)&lA[1][0]};
  const unsigned int ldsB[2] = {(unsigned int)(size_t)&lB[0][0],
                                (unsigned int)(size_t)&lB[1][0]};

  v8f acc[2][2];
  acc[0][0] = v8f{}; acc[0][1] = v8f{}; acc[1][0] = v8f{}; acc[1][1] = v8f{};

  const int nk = K / 32;
  if (tid == 0) {   // TDM issues once per wave; only wave 0 / lane 0 path issues
    tdm_load_2d(ldsA[0], &A[(size_t)bm * K], 32, 128, K, M);
    tdm_load_2d(ldsB[0], &W[(size_t)bn * K], 32, 64, K, N);
  }

  for (int i = 0; i < nk; ++i) {
    const int cur = i & 1;
    __syncthreads();              // all waves done reading buffer we overwrite next
    if (tid == 0) {
      if (i + 1 < nk) {
        tdm_load_2d(ldsA[cur ^ 1], &A[(size_t)bm * K + (size_t)(i + 1) * 32],
                    32, 128, K, M);
        tdm_load_2d(ldsB[cur ^ 1], &W[(size_t)bn * K + (size_t)(i + 1) * 32],
                    32, 64, K, N);
        __builtin_amdgcn_s_wait_tensorcnt(2);  // current 2 TDM ops done (in-order)
      } else {
        __builtin_amdgcn_s_wait_tensorcnt(0);
      }
    }
    __syncthreads();              // current buffer visible to all waves

    Frag16 fa0, fa1;  // A 16x32 frags: elem e -> k = kh*8 + (e&7) + (e>>3)*16
    fa0.u[0] = *(const uint4*)&lA[cur][(wm * 32 + am) * 32 + kh * 8];
    fa0.u[1] = *(const uint4*)&lA[cur][(wm * 32 + am) * 32 + kh * 8 + 16];
    fa1.u[0] = *(const uint4*)&lA[cur][(wm * 32 + 16 + am) * 32 + kh * 8];
    fa1.u[1] = *(const uint4*)&lA[cur][(wm * 32 + 16 + am) * 32 + kh * 8 + 16];

    Frag16 fb0, fb1;  // B 32x16 frags: elem e -> k = half*8 + (e&7) + (e>>3)*16
    const int nb0  = wn * 32 + cn;
    const int koff = half * 8;
    fb0.u[0] = *(const uint4*)&lB[cur][nb0 * 32 + koff];
    fb0.u[1] = *(const uint4*)&lB[cur][nb0 * 32 + koff + 16];
    fb1.u[0] = *(const uint4*)&lB[cur][(nb0 + 16) * 32 + koff];
    fb1.u[1] = *(const uint4*)&lB[cur][(nb0 + 16) * 32 + koff + 16];

    acc[0][0] = wmma_bf16(fa0.v, fb0.v, acc[0][0]);
    acc[0][1] = wmma_bf16(fa0.v, fb1.v, acc[0][1]);
    acc[1][0] = wmma_bf16(fa1.v, fb0.v, acc[1][0]);
    acc[1][1] = wmma_bf16(fa1.v, fb1.v, acc[1][1]);
  }

  // C layout: vgpr r holds row m = r + 8*half, col = cn
  for (int mi = 0; mi < 2; ++mi)
    for (int ni = 0; ni < 2; ++ni)
      for (int r = 0; r < 8; ++r) {
        const size_t row = (size_t)(bm + wm * 32 + mi * 16 + half * 8 + r);
        const size_t col = (size_t)(bn + wn * 32 + ni * 16 + cn);
        const float v = acc[mi][ni][r];
        if (EPI == 0) {
          ((bf16_t*)Out)[row * N + col] = (bf16_t)v;
        } else if (EPI == 1) {
          ((float*)Out)[row * N + col] = v + Res[row * N + col];
        } else {
          ((float*)Out)[row * N + col] += v;
        }
      }
}

// ---------------------------------------------------------------------------
// RoPE in place on [B,T,NH,HD] bf16; freqs_cis is [T, HD/2, 2] fp32
// ---------------------------------------------------------------------------
__global__ void rope_bf16(bf16_t* __restrict__ t, const float* __restrict__ fc,
                          int NH, size_t npairs) {
  size_t p = (size_t)blockIdx.x * blockDim.x + threadIdx.x;
  if (p >= npairs) return;
  const int i   = (int)(p & (HD_ / 2 - 1));
  const size_t r = p >> 6;                 // HD/2 == 64
  const int tt  = (int)((r / NH) % T_);
  const float c = fc[((size_t)tt * (HD_ / 2) + i) * 2 + 0];
  const float s = fc[((size_t)tt * (HD_ / 2) + i) * 2 + 1];
  const float a = (float)t[2 * p];
  const float b = (float)t[2 * p + 1];
  t[2 * p]     = (bf16_t)(a * c - b * s);
  t[2 * p + 1] = (bf16_t)(a * s + b * c);
}

// ---------------------------------------------------------------------------
// V transpose: [b,t,kv,hd] -> [b,kv,hd,t] (bf16), 32x32 LDS tiles
// ---------------------------------------------------------------------------
__global__ __launch_bounds__(256) void transpose_v(const bf16_t* __restrict__ v,
                                                   bf16_t* __restrict__ vt) {
  __shared__ bf16_t tile[32][33];
  const int bkv = blockIdx.z;                 // b*KV + kv
  const int t0  = blockIdx.x * 32;
  const int d0  = blockIdx.y * 32;
  const int tx = threadIdx.x, ty = threadIdx.y;   // block (32,8)
  const int b = bkv / KV_, kv = bkv % KV_;
  for (int i = ty; i < 32; i += 8)
    tile[i][tx] = v[(((size_t)b * T_ + t0 + i) * KV_ + kv) * HD_ + d0 + tx];
  __syncthreads();
  for (int i = ty; i < 32; i += 8)
    vt[((size_t)bkv * HD_ + d0 + i) * T_ + t0 + tx] = tile[tx][i];
}

// ---------------------------------------------------------------------------
// Flash attention: 1 wave per (b, h, 16-query tile). Causal, online softmax.
// Q [B,T,H,HD] bf16 (rope'd), K [B,T,KV,HD] bf16 (rope'd), Vt [B,KV,HD,T] bf16.
// Y [B,T,H,HD] bf16.
// ---------------------------------------------------------------------------
__global__ __launch_bounds__(32) void attn_wmma(const bf16_t* __restrict__ q,
                                                const bf16_t* __restrict__ k,
                                                const bf16_t* __restrict__ vt,
                                                bf16_t* __restrict__ y) {
  __shared__ __align__(16) bf16_t lP[16 * 32];

  const int lane = threadIdx.x;
  const int qt = blockIdx.x, h = blockIdx.y, b = blockIdx.z;
  const int kvh = h / REP_;
  const int q0  = qt * 16;
  const int am = lane & 15, kh = lane >> 4;
  const int cn = lane & 15, half = lane >> 4;

  // Q fragments for all 4 K-chunks of HD=128 (A layout)
  Frag16 fq[4];
  const bf16_t* qrow = q + (((size_t)b * T_ + q0 + am) * H_ + h) * HD_;
  for (int c = 0; c < 4; ++c) {
    fq[c].u[0] = *(const uint4*)&qrow[c * 32 + kh * 8];
    fq[c].u[1] = *(const uint4*)&qrow[c * 32 + kh * 8 + 16];
  }

  const bf16_t* vtbase = vt + (size_t)(b * KV_ + kvh) * HD_ * T_;

  v8f o[8];
  float m_i[8], l_i[8], alph[8];
  for (int nt = 0; nt < 8; ++nt) o[nt] = v8f{};
  for (int r = 0; r < 8; ++r) { m_i[r] = -3e38f; l_i[r] = 0.f; }

  const float sc = 0.08838834764831845f * 1.4426950408889634f;  // 1/sqrt(HD)*log2e
  const int nkb = (q0 + 16 + 31) / 32;

  for (int kb = 0; kb < nkb; ++kb) {
    const int kbase = kb * 32;

    // ---- scores: two 16x16 tiles (keys kbase..+15 and +16..+31) ----
    v8f s0 = {}, s1 = {};
    const bf16_t* krow0 = k + (((size_t)b * T_ + kbase + cn) * KV_ + kvh) * HD_;
    const bf16_t* krow1 = krow0 + (size_t)16 * KV_ * HD_;
    for (int c = 0; c < 4; ++c) {
      Frag16 fb0, fb1;
      const int ko = c * 32 + half * 8;   // B layout: lane=key, k = half*8 + (e&7) + (e>>3)*16
      fb0.u[0] = *(const uint4*)&krow0[ko];
      fb0.u[1] = *(const uint4*)&krow0[ko + 16];
      fb1.u[0] = *(const uint4*)&krow1[ko];
      fb1.u[1] = *(const uint4*)&krow1[ko + 16];
      s0 = wmma_bf16(fq[c].v, fb0.v, s0);
      s1 = wmma_bf16(fq[c].v, fb1.v, s1);
    }

    // ---- online softmax update (rows striped per C layout) ----
    const bool maskblk = (kb == nkb - 1);
    for (int r = 0; r < 8; ++r) {
      float a0 = s0[r] * sc, a1 = s1[r] * sc;
      if (maskblk) {
        const int qidx = q0 + r + 8 * half;
        if (kbase + cn > qidx)      a0 = -3e38f;
        if (kbase + cn + 16 > qidx) a1 = -3e38f;
      }
      float mx = half16_max(fmaxf(a0, a1));
      const float nm = fmaxf(m_i[r], mx);
      const float al = exp2f(m_i[r] - nm);
      const float p0 = exp2f(a0 - nm);
      const float p1 = exp2f(a1 - nm);
      const float rs = half16_sum(p0 + p1);
      l_i[r] = l_i[r] * al + rs;
      m_i[r] = nm;
      alph[r] = al;
      // restage P (16 rows x 32 keys) row-major for A-fragment reload
      lP[(r + 8 * half) * 32 + cn]      = (bf16_t)p0;
      lP[(r + 8 * half) * 32 + cn + 16] = (bf16_t)p1;
    }
    for (int nt = 0; nt < 8; ++nt)
      for (int r = 0; r < 8; ++r) o[nt][r] *= alph[r];
    __syncthreads();

    // ---- PV: P(16x32) x V(32x128) -> 8 accum tiles ----
    Frag16 fp;
    fp.u[0] = *(const uint4*)&lP[am * 32 + kh * 8];
    fp.u[1] = *(const uint4*)&lP[am * 32 + kh * 8 + 16];
    for (int nt = 0; nt < 8; ++nt) {
      Frag16 fv;  // B layout from transposed V: lane=hd col, keys half*8 + (e&7) + (e>>3)*16
      const bf16_t* col = vtbase + (size_t)(nt * 16 + cn) * T_ + kbase + half * 8;
      fv.u[0] = *(const uint4*)col;
      fv.u[1] = *(const uint4*)(col + 16);
      o[nt] = wmma_bf16(fp.v, fv.v, o[nt]);
    }
    __syncthreads();
  }

  // ---- normalize + store Y[b, q, h, :] ----
  for (int nt = 0; nt < 8; ++nt)
    for (int r = 0; r < 8; ++r) {
      const int m = r + 8 * half;
      const size_t row = ((size_t)b * T_ + q0 + m) * H_ + h;
      y[row * HD_ + nt * 16 + cn] = (bf16_t)(o[nt][r] / l_i[r]);
    }
}

// ---------------------------------------------------------------------------
// u = silu(a) * b  (in place into a)
// ---------------------------------------------------------------------------
__global__ void silu_mul(bf16_t* __restrict__ a, const bf16_t* __restrict__ b,
                         size_t n) {
  size_t i = (size_t)blockIdx.x * blockDim.x + threadIdx.x;
  if (i < n) {
    const float x = (float)a[i];
    const float s = x / (1.f + __expf(-x));
    a[i] = (bf16_t)(s * (float)b[i]);
  }
}

// ---------------------------------------------------------------------------
extern "C" void kernel_launch(void* const* d_in, const int* in_sizes, int n_in,
                              void* d_out, int out_size, void* d_ws, size_t ws_size,
                              hipStream_t stream) {
  const float* x  = (const float*)d_in[0];
  const float* fc = (const float*)d_in[1];
  const float* wq = (const float*)d_in[2];
  const float* wk = (const float*)d_in[3];
  const float* wv = (const float*)d_in[4];
  const float* wo = (const float*)d_in[5];
  const float* w1 = (const float*)d_in[6];
  const float* w2 = (const float*)d_in[7];
  const float* cp = (const float*)d_in[8];
  const float* g1 = (const float*)d_in[9];
  const float* g2 = (const float*)d_in[10];
  float* out = (float*)d_out;

  size_t off = 0;
  auto carve = [&](size_t bytes) {
    void* p = (char*)d_ws + off;
    off += (bytes + 255) & ~(size_t)255;
    return p;
  };
  bf16_t* h   = (bf16_t*)carve((size_t)BT_ * C_ * 2);     // reused as h2
  bf16_t* wqb = (bf16_t*)carve((size_t)C_ * C_ * 2);
  bf16_t* wkb = (bf16_t*)carve((size_t)KV_ * HD_ * C_ * 2);
  bf16_t* wvb = (bf16_t*)carve((size_t)KV_ * HD_ * C_ * 2);
  bf16_t* wob = (bf16_t*)carve((size_t)C_ * C_ * 2);
  bf16_t* w1b = (bf16_t*)carve((size_t)FFN_ * C_ * 2);
  bf16_t* w2b = (bf16_t*)carve((size_t)FFN_ * C_ * 2);
  bf16_t* cpb = (bf16_t*)carve((size_t)C_ * FFN_ * 2);
  bf16_t* qb  = (bf16_t*)carve((size_t)BT_ * C_ * 2);
  bf16_t* kb  = (bf16_t*)carve((size_t)BT_ * KV_ * HD_ * 2);
  bf16_t* vb  = (bf16_t*)carve((size_t)BT_ * KV_ * HD_ * 2);
  bf16_t* vtb = (bf16_t*)carve((size_t)BT_ * KV_ * HD_ * 2);
  bf16_t* yb  = (bf16_t*)carve((size_t)BT_ * C_ * 2);
  bf16_t* t1  = (bf16_t*)carve((size_t)BT_ * FFN_ * 2);
  bf16_t* t2  = (bf16_t*)carve((size_t)BT_ * FFN_ * 2);

  auto cvt = [&](const float* src, bf16_t* dst, size_t n) {
    cvt_f32_bf16<<<(unsigned)((n + 255) / 256), 256, 0, stream>>>(src, dst, n);
  };
  cvt(wq, wqb, (size_t)C_ * C_);
  cvt(wk, wkb, (size_t)KV_ * HD_ * C_);
  cvt(wv, wvb, (size_t)KV_ * HD_ * C_);
  cvt(wo, wob, (size_t)C_ * C_);
  cvt(w1, w1b, (size_t)FFN_ * C_);
  cvt(w2, w2b, (size_t)FFN_ * C_);
  cvt(cp, cpb, (size_t)C_ * FFN_);

  // h = rmsnorm(x, g1)
  rmsnorm_bf16<<<BT_, 256, 0, stream>>>(x, g1, h);

  // q/k/v projections
  gemm_bf16_wmma<0><<<dim3(C_ / 64, BT_ / 128), 256, 0, stream>>>(
      h, wqb, qb, nullptr, BT_, C_, C_);
  gemm_bf16_wmma<0><<<dim3(KV_ * HD_ / 64, BT_ / 128), 256, 0, stream>>>(
      h, wkb, kb, nullptr, BT_, KV_ * HD_, C_);
  gemm_bf16_wmma<0><<<dim3(KV_ * HD_ / 64, BT_ / 128), 256, 0, stream>>>(
      h, wvb, vb, nullptr, BT_, KV_ * HD_, C_);

  // RoPE
  {
    size_t npq = (size_t)BT_ * H_ * (HD_ / 2);
    rope_bf16<<<(unsigned)((npq + 255) / 256), 256, 0, stream>>>(qb, fc, H_, npq);
    size_t npk = (size_t)BT_ * KV_ * (HD_ / 2);
    rope_bf16<<<(unsigned)((npk + 255) / 256), 256, 0, stream>>>(kb, fc, KV_, npk);
  }

  // V transpose for PV fragment loads
  transpose_v<<<dim3(T_ / 32, HD_ / 32, B_ * KV_), dim3(32, 8), 0, stream>>>(vb, vtb);

  // flash attention
  attn_wmma<<<dim3(T_ / 16, H_, B_), 32, 0, stream>>>(qb, kb, vtb, yb);

  // x1 = x + y @ wo^T   -> d_out
  gemm_bf16_wmma<1><<<dim3(C_ / 64, BT_ / 128), 256, 0, stream>>>(
      yb, wob, out, x, BT_, C_, C_);

  // h2 = rmsnorm(x1, g2)  (reuse h)
  rmsnorm_bf16<<<BT_, 256, 0, stream>>>(out, g2, h);

  // MLP
  gemm_bf16_wmma<0><<<dim3(FFN_ / 64, BT_ / 128), 256, 0, stream>>>(
      h, w1b, t1, nullptr, BT_, FFN_, C_);
  gemm_bf16_wmma<0><<<dim3(FFN_ / 64, BT_ / 128), 256, 0, stream>>>(
      h, w2b, t2, nullptr, BT_, FFN_, C_);
  {
    size_t n = (size_t)BT_ * FFN_;
    silu_mul<<<(unsigned)((n + 255) / 256), 256, 0, stream>>>(t1, t2, n);
  }
  // d_out += u @ c_proj^T
  gemm_bf16_wmma<2><<<dim3(C_ / 64, BT_ / 128), 256, 0, stream>>>(
      t1, cpb, out, nullptr, BT_, C_, FFN_);
}
